// MACE_56092272885860
// MI455X (gfx1250) — matrix operand order repaired and checked by reading, not compiled, not run on hardware
//
#include <hip/hip_runtime.h>
#include <hip/hip_bf16.h>
#include <math.h>

typedef __attribute__((ext_vector_type(16))) _Float16 v16h;
typedef __attribute__((ext_vector_type(8)))  _Float16 v8h;
typedef __attribute__((ext_vector_type(8)))  float    v8f;
typedef __attribute__((ext_vector_type(4)))  unsigned int u32x4;
typedef __attribute__((ext_vector_type(8)))  int i32x8;
typedef __attribute__((ext_vector_type(4)))  int i32x4;

__device__ __forceinline__ v8f zero8() {
    v8f z;
#pragma unroll
    for (int i = 0; i < 8; ++i) z[i] = 0.0f;
    return z;
}

__device__ __forceinline__ v8f wmma16x16x32(v16h a, v16h b, v8f c) {
    // (neg_a, A, neg_b, B, c_mod, C, reuse_a, reuse_b)
    return __builtin_amdgcn_wmma_f32_16x16x32_f16(false, a, false, b, (short)0, c,
                                                  false, false);
}

// A-fragment (16x32 f16, M x K) from row-major LDS tile [m][k], row stride lda halves.
__device__ __forceinline__ v16h ldsA_frag(const _Float16* base, int lane, int lda, int koff) {
    const int m  = lane & 15;
    const int kb = (lane < 16) ? 0 : 16;
    const _Float16* p = base + m * lda + koff + kb;
    v8h lo = *(const v8h*)(p);
    v8h hi = *(const v8h*)(p + 8);
    v16h r;
#pragma unroll
    for (int i = 0; i < 8; ++i) { r[i] = lo[i]; r[8 + i] = hi[i]; }
    return r;
}

// B-fragment (32x16 f16, K x N) from TRANSPOSED LDS weights [n][k], row stride ldk halves.
__device__ __forceinline__ v16h ldsB_fragT(const _Float16* baseT, int lane, int ldk, int koff) {
    const int n  = lane & 15;
    const int kb = (lane < 16) ? 0 : 16;
    return *(const v16h*)(baseT + n * ldk + koff + kb);
}

__device__ __forceinline__ float silu(float v) {
    return v / (1.0f + __expf(-v));
}

// ---------------------------------------------------------------------------
// TDM 1-D copy: n f16 elements, global -> LDS, one descriptor.
// D# group0: count=1 | lds_addr | global_addr[56:0] | type=2
// D# group1: wg_mask=0, data_size=1(2B), tensor_dim0=tile_dim0=n, tensor_dim1=1
// ---------------------------------------------------------------------------
__device__ __forceinline__ void tdm_load_1d(const _Float16* gptr, _Float16* lptr,
                                            unsigned n_elems) {
    const unsigned lds = (unsigned)(unsigned long long)lptr;   // addr[31:0] = LDS offset
    const unsigned long long ga = (unsigned long long)gptr;
    u32x4 g0;
    g0[0] = 1u;                                        // count=1, user mode
    g0[1] = lds;                                       // lds_addr
    g0[2] = (unsigned)(ga & 0xffffffffull);            // global_addr[31:0]
    g0[3] = (unsigned)((ga >> 32) & 0x1ffffffull) |    // global_addr[56:32]
            (2u << 30);                                // type = 2 ("image")
    i32x8 g1;
    g1[0] = (int)(1u << 16);                           // data_size=1 (2 bytes)
    g1[1] = (int)((n_elems & 0xffffu) << 16);          // tensor_dim0[15:0]
    g1[2] = (int)((n_elems >> 16) | (1u << 16));       // tensor_dim0[31:16], tensor_dim1=1
    g1[3] = (int)((n_elems & 0xffffu) << 16);          // tile_dim0 = n
    g1[4] = 0;                                         // tile_dim1=0, tile_dim2=0
    g1[5] = (int)n_elems;                              // tensor_dim0_stride
    g1[6] = 0;
    g1[7] = 0;
    i32x4 z4 = {0, 0, 0, 0};
#if __clang_major__ >= 23
    i32x8 z8 = {0, 0, 0, 0, 0, 0, 0, 0};
    __builtin_amdgcn_tensor_load_to_lds(g0, g1, z4, z4, z8, 0);
#else
    __builtin_amdgcn_tensor_load_to_lds(g0, g1, z4, z4, 0);
#endif
}

// ---------------------------------------------------------------------------
// Weight prep: f32 -> f16, transposed to [n][k], msg_w1 K padded 136 -> 160.
// ---------------------------------------------------------------------------
__global__ void prep_weights_kernel(const float* __restrict__ mw1,
                                    const float* __restrict__ mw2,
                                    const float* __restrict__ uw1,
                                    const float* __restrict__ uw2,
                                    const float* __restrict__ ew1,
                                    _Float16* __restrict__ wh, int NL) {
    const int i = blockIdx.x * blockDim.x + threadIdx.x;
    const int A = NL * 10240, B = NL * 4096, C = NL * 8192, D = NL * 4096;
    float v;
    if (i < A) {
        int l = i / 10240, rem = i % 10240, n = rem / 160, k = rem % 160;
        v = (k < 136) ? mw1[(l * 136 + k) * 64 + n] : 0.0f;
    } else if (i < A + B) {
        int j = i - A; int l = j / 4096, rem = j % 4096, n = rem / 64, k = rem % 64;
        v = mw2[(l * 64 + k) * 64 + n];
    } else if (i < A + B + C) {
        int j = i - A - B; int l = j / 8192, rem = j % 8192, n = rem / 128, k = rem % 128;
        v = uw1[(l * 128 + k) * 64 + n];
    } else if (i < A + B + C + D) {
        int j = i - A - B - C; int l = j / 4096, rem = j % 4096, n = rem / 64, k = rem % 64;
        v = uw2[(l * 64 + k) * 64 + n];
    } else if (i < A + B + C + D + 4096) {
        int j = i - A - B - C - D; int n = j / 64, k = j % 64;
        v = ew1[k * 64 + n];
    } else {
        return;
    }
    wh[i] = (_Float16)v;
}

__global__ void edge_attr_kernel(const float* __restrict__ pos,
                                 const int* __restrict__ erow,
                                 const int* __restrict__ ecol,
                                 float* __restrict__ ea, int E) {
    const int e = blockIdx.x * blockDim.x + threadIdx.x;
    if (e >= E) return;
    const int r = erow[e], c = ecol[e];
    const float dx = pos[r * 3 + 0] - pos[c * 3 + 0];
    const float dy = pos[r * 3 + 1] - pos[c * 3 + 1];
    const float dz = pos[r * 3 + 2] - pos[c * 3 + 2];
    const float d  = sqrtf(dx * dx + dy * dy + dz * dz);
    float env = 0.0f;
    if (d < 5.0f) {
        const float ce = __cosf(d * 0.31415926535897932f);  // pi/(2*cutoff)
        env = ce * ce;
    }
    const float ds = (d > 0.0f) ? d : 1.0f;
    const float sc = env / ds;
#pragma unroll
    for (int b = 0; b < 8; ++b)
        ea[e * 8 + b] = __sinf((float)(b + 1) * 0.62831853071795865f * d) * sc;
}

__global__ void gather_embed_kernel(const int* __restrict__ z,
                                    const float* __restrict__ embed,
                                    float* __restrict__ x, int N) {
    const int i = blockIdx.x * blockDim.x + threadIdx.x;
    if (i >= N * 64) return;
    const int n = i >> 6, h = i & 63;
    x[i] = embed[z[n] * 64 + h];
}

__global__ void zero_kernel(float* __restrict__ p, int n) {
    const int i = blockIdx.x * blockDim.x + threadIdx.x;
    if (i < n) p[i] = 0.0f;
}

// ---------------------------------------------------------------------------
// Message + scatter-add. 256 edges/block as 4 tiles of 64; 128 threads (4 waves).
// Weights staged once per block via TDM (wave 0), amortized over 4 tiles.
// ---------------------------------------------------------------------------
__global__ __launch_bounds__(128) void msg_kernel(
    const float* __restrict__ x, const float* __restrict__ ea,
    const int* __restrict__ erow, const int* __restrict__ ecol,
    const _Float16* __restrict__ w1T, const float* __restrict__ b1,
    const _Float16* __restrict__ w2T, const float* __restrict__ b2,
    float* __restrict__ agg, int E) {
    __shared__ __attribute__((aligned(32))) _Float16 sA[64 * 160];
    __shared__ __attribute__((aligned(32))) _Float16 sW1[64 * 160];
    __shared__ __attribute__((aligned(32))) _Float16 sW2[64 * 64];
    __shared__ __attribute__((aligned(32))) _Float16 sH[64 * 64];
    __shared__ int sRow[64];
    __shared__ int sCol[64];

    const int t    = threadIdx.x;
    const int lane = t & 31;
    const int wave = t >> 5;
    const int nloc = lane & 15;
    const int moff = (lane < 16) ? 0 : 8;

    if (wave == 0) {                       // async weight staging via Tensor Data Mover
        tdm_load_1d(w1T, sW1, 64 * 160);
        tdm_load_1d(w2T, sW2, 64 * 64);
    }

    for (int tile = 0; tile < 4; ++tile) {
        const int e0 = (blockIdx.x * 4 + tile) * 64;
        if (tile) __syncthreads();         // prior tile done reading sA/sCol
        if (t < 64) {
            const int e = e0 + t;
            sRow[t] = (e < E) ? erow[e] : -1;
            sCol[t] = (e < E) ? ecol[e] : -1;
        }
        __syncthreads();

        // Build msg_in tile: [x[col] | x[row] | edge_attr | 0-pad]
        for (int i = 0; i < 80; ++i) {
            const int idx = t + i * 128;
            const int el = idx / 160, k = idx % 160;
            const int r = sRow[el], c = sCol[el];
            float v = 0.0f;
            if (c >= 0) {
                if (k < 64)        v = x[c * 64 + k];
                else if (k < 128)  v = x[r * 64 + (k - 64)];
                else if (k < 136)  v = ea[(e0 + el) * 8 + (k - 128)];
            }
            sA[idx] = (_Float16)v;
        }
        if (tile == 0 && wave == 0) __builtin_amdgcn_s_wait_tensorcnt(0);
        __syncthreads();

        // GEMM1: (16x160) x (160x64) per wave -> 5 k-steps x 4 n-tiles
        v8f acc[4];
#pragma unroll
        for (int j = 0; j < 4; ++j) acc[j] = zero8();
#pragma unroll
        for (int kt = 0; kt < 5; ++kt) {
            v16h a = ldsA_frag(sA + wave * 16 * 160, lane, 160, kt * 32);
#pragma unroll
            for (int j = 0; j < 4; ++j) {
                v16h b = ldsB_fragT(sW1 + j * 16 * 160, lane, 160, kt * 32);
                acc[j] = wmma16x16x32(a, b, acc[j]);
            }
        }

#pragma unroll
        for (int j = 0; j < 4; ++j) {
            const float bb = b1[j * 16 + nloc];
#pragma unroll
            for (int r = 0; r < 8; ++r) {
                const float v = silu(acc[j][r] + bb);
                sH[(wave * 16 + moff + r) * 64 + j * 16 + nloc] = (_Float16)v;
            }
        }
        // sH rows are produced and consumed by the same wave; DScnt ordering suffices.

        // GEMM2: (16x64) x (64x64) -> 2 k-steps x 4 n-tiles
        v8f acc2[4];
#pragma unroll
        for (int j = 0; j < 4; ++j) acc2[j] = zero8();
#pragma unroll
        for (int kt = 0; kt < 2; ++kt) {
            v16h a = ldsA_frag(sH + wave * 16 * 64, lane, 64, kt * 32);
#pragma unroll
            for (int j = 0; j < 4; ++j) {
                v16h b = ldsB_fragT(sW2 + j * 16 * 64, lane, 64, kt * 32);
                acc2[j] = wmma16x16x32(a, b, acc2[j]);
            }
        }

        int cc[8];
#pragma unroll
        for (int r = 0; r < 8; ++r) cc[r] = sCol[wave * 16 + moff + r];
#pragma unroll
        for (int j = 0; j < 4; ++j) {
            const float bb = b2[j * 16 + nloc];
#pragma unroll
            for (int r = 0; r < 8; ++r) {
                if (cc[r] >= 0)
                    atomicAdd(&agg[cc[r] * 64 + j * 16 + nloc], acc2[j][r] + bb);
            }
        }
    }
}

// ---------------------------------------------------------------------------
// Node update: u_in(64x128) @ W1T -> SiLU -> @ W2T ; x += upd. 64 nodes/block.
// ---------------------------------------------------------------------------
__global__ __launch_bounds__(128) void upd_kernel(
    float* __restrict__ x, const float* __restrict__ agg,
    const _Float16* __restrict__ w1T, const float* __restrict__ b1,
    const _Float16* __restrict__ w2T, const float* __restrict__ b2, int N) {
    __shared__ __attribute__((aligned(32))) _Float16 sU[64 * 128];
    __shared__ __attribute__((aligned(32))) _Float16 sW1[64 * 128];
    __shared__ __attribute__((aligned(32))) _Float16 sW2[64 * 64];
    __shared__ __attribute__((aligned(32))) _Float16 sH[64 * 64];

    const int t    = threadIdx.x;
    const int lane = t & 31;
    const int wave = t >> 5;
    const int n0   = blockIdx.x * 64;

    if (wave == 0) {
        tdm_load_1d(w1T, sW1, 64 * 128);
        tdm_load_1d(w2T, sW2, 64 * 64);
    }
    for (int i = 0; i < 64; ++i) {
        const int idx = t + i * 128;
        const int nl = idx >> 7, k = idx & 127;
        const int n = n0 + nl;
        float v = 0.0f;
        if (n < N) v = (k < 64) ? x[n * 64 + k] : agg[n * 64 + (k - 64)];
        sU[idx] = (_Float16)v;
    }
    if (wave == 0) __builtin_amdgcn_s_wait_tensorcnt(0);
    __syncthreads();

    v8f acc[4];
#pragma unroll
    for (int j = 0; j < 4; ++j) acc[j] = zero8();
#pragma unroll
    for (int kt = 0; kt < 4; ++kt) {
        v16h a = ldsA_frag(sU + wave * 16 * 128, lane, 128, kt * 32);
#pragma unroll
        for (int j = 0; j < 4; ++j) {
            v16h b = ldsB_fragT(sW1 + j * 16 * 128, lane, 128, kt * 32);
            acc[j] = wmma16x16x32(a, b, acc[j]);
        }
    }

    const int nloc = lane & 15;
    const int moff = (lane < 16) ? 0 : 8;
#pragma unroll
    for (int j = 0; j < 4; ++j) {
        const float bb = b1[j * 16 + nloc];
#pragma unroll
        for (int r = 0; r < 8; ++r) {
            const float v = silu(acc[j][r] + bb);
            sH[(wave * 16 + moff + r) * 64 + j * 16 + nloc] = (_Float16)v;
        }
    }

    v8f acc2[4];
#pragma unroll
    for (int j = 0; j < 4; ++j) acc2[j] = zero8();
#pragma unroll
    for (int kt = 0; kt < 2; ++kt) {
        v16h a = ldsA_frag(sH + wave * 16 * 64, lane, 64, kt * 32);
#pragma unroll
        for (int j = 0; j < 4; ++j) {
            v16h b = ldsB_fragT(sW2 + j * 16 * 64, lane, 64, kt * 32);
            acc2[j] = wmma16x16x32(a, b, acc2[j]);
        }
    }

#pragma unroll
    for (int j = 0; j < 4; ++j) {
        const float bb = b2[j * 16 + nloc];
#pragma unroll
        for (int r = 0; r < 8; ++r) {
            const int n = n0 + wave * 16 + moff + r;
            if (n < N) x[n * 64 + j * 16 + nloc] += acc2[j][r] + bb;
        }
    }
}

// ---------------------------------------------------------------------------
// Readout: silu(x @ eh_w1 + b1) . eh_w2 + b2 ; segment-sum into out[batch[n]].
// ---------------------------------------------------------------------------
__global__ __launch_bounds__(128) void readout_kernel(
    const float* __restrict__ x, const _Float16* __restrict__ w1T,
    const float* __restrict__ b1, const float* __restrict__ w2v,
    const float* __restrict__ b2, const int* __restrict__ batch,
    float* __restrict__ out, int N) {
    __shared__ __attribute__((aligned(32))) _Float16 sX[64 * 64];
    __shared__ __attribute__((aligned(32))) _Float16 sW[64 * 64];
    __shared__ float sHf[64 * 64];
    __shared__ float sV[64];
    __shared__ float sP[128];

    const int t    = threadIdx.x;
    const int lane = t & 31;
    const int wave = t >> 5;
    const int n0   = blockIdx.x * 64;

    if (wave == 0) tdm_load_1d(w1T, sW, 64 * 64);
    for (int i = 0; i < 32; ++i) {
        const int idx = t + i * 128;
        const int nl = idx >> 6, k = idx & 63;
        const int n = n0 + nl;
        sX[idx] = (_Float16)((n < N) ? x[n * 64 + k] : 0.0f);
    }
    if (t < 64) sV[t] = w2v[t];
    if (wave == 0) __builtin_amdgcn_s_wait_tensorcnt(0);
    __syncthreads();

    v8f acc[4];
#pragma unroll
    for (int j = 0; j < 4; ++j) acc[j] = zero8();
#pragma unroll
    for (int kt = 0; kt < 2; ++kt) {
        v16h a = ldsA_frag(sX + wave * 16 * 64, lane, 64, kt * 32);
#pragma unroll
        for (int j = 0; j < 4; ++j) {
            v16h b = ldsB_fragT(sW + j * 16 * 64, lane, 64, kt * 32);
            acc[j] = wmma16x16x32(a, b, acc[j]);
        }
    }

    const int nloc = lane & 15;
    const int moff = (lane < 16) ? 0 : 8;
#pragma unroll
    for (int j = 0; j < 4; ++j) {
        const float bb = b1[j * 16 + nloc];
#pragma unroll
        for (int r = 0; r < 8; ++r)
            sHf[(wave * 16 + moff + r) * 64 + j * 16 + nloc] = silu(acc[j][r] + bb);
    }
    __syncthreads();

    const int nl = t >> 1, half = (t & 1) * 32;
    float s = 0.0f;
#pragma unroll
    for (int j = 0; j < 32; ++j) s += sHf[nl * 64 + half + j] * sV[half + j];
    sP[t] = s;
    __syncthreads();

    if (t < 64) {
        const int n = n0 + t;
        if (n < N) atomicAdd(&out[batch[n]], sP[2 * t] + sP[2 * t + 1] + b2[0]);
    }
}

// ---------------------------------------------------------------------------
extern "C" void kernel_launch(void* const* d_in, const int* in_sizes, int n_in,
                              void* d_out, int out_size, void* d_ws, size_t ws_size,
                              hipStream_t stream) {
    const int N  = in_sizes[0];
    const int E  = in_sizes[2] / 2;
    const int NL = in_sizes[6] / 64;
    const int NG = out_size;

    const int*   z      = (const int*)d_in[0];
    const float* pos    = (const float*)d_in[1];
    const int*   eidx   = (const int*)d_in[2];
    const int*   batch  = (const int*)d_in[3];
    const float* embed  = (const float*)d_in[4];
    const float* msg_w1 = (const float*)d_in[5];
    const float* msg_b1 = (const float*)d_in[6];
    const float* msg_w2 = (const float*)d_in[7];
    const float* msg_b2 = (const float*)d_in[8];
    const float* upd_w1 = (const float*)d_in[9];
    const float* upd_b1 = (const float*)d_in[10];
    const float* upd_w2 = (const float*)d_in[11];
    const float* upd_b2 = (const float*)d_in[12];
    const float* eh_w1  = (const float*)d_in[13];
    const float* eh_b1  = (const float*)d_in[14];
    const float* eh_w2  = (const float*)d_in[15];
    const float* eh_b2  = (const float*)d_in[16];
    float* out = (float*)d_out;

    char* ws = (char*)d_ws;
    size_t off = 0;
    float* ea = (float*)(ws + off);  off += (size_t)E * 8 * sizeof(float);
    off = (off + 255) & ~(size_t)255;
    float* xbuf = (float*)(ws + off); off += (size_t)N * 64 * sizeof(float);
    off = (off + 255) & ~(size_t)255;
    float* agg = (float*)(ws + off);  off += (size_t)N * 64 * sizeof(float);
    off = (off + 255) & ~(size_t)255;
    _Float16* wh = (_Float16*)(ws + off);

    _Float16* whMsg1 = wh;                          // [NL][64][160]
    _Float16* whMsg2 = whMsg1 + (size_t)NL * 10240; // [NL][64][64]
    _Float16* whUpd1 = whMsg2 + (size_t)NL * 4096;  // [NL][64][128]
    _Float16* whUpd2 = whUpd1 + (size_t)NL * 8192;  // [NL][64][64]
    _Float16* whEh1  = whUpd2 + (size_t)NL * 4096;  // [64][64]

    const int* erow = eidx;
    const int* ecol = eidx + E;

    const int whTotal = NL * 26624 + 4096;
    prep_weights_kernel<<<(whTotal + 255) / 256, 256, 0, stream>>>(
        msg_w1, msg_w2, upd_w1, upd_w2, eh_w1, wh, NL);

    edge_attr_kernel<<<(E + 255) / 256, 256, 0, stream>>>(pos, erow, ecol, ea, E);
    gather_embed_kernel<<<(N * 64 + 255) / 256, 256, 0, stream>>>(z, embed, xbuf, N);

    for (int l = 0; l < NL; ++l) {
        zero_kernel<<<(N * 64 + 255) / 256, 256, 0, stream>>>(agg, N * 64);
        msg_kernel<<<(E + 255) / 256, 128, 0, stream>>>(
            xbuf, ea, erow, ecol,
            whMsg1 + (size_t)l * 10240, msg_b1 + (size_t)l * 64,
            whMsg2 + (size_t)l * 4096,  msg_b2 + (size_t)l * 64, agg, E);
        upd_kernel<<<(N + 63) / 64, 128, 0, stream>>>(
            xbuf, agg,
            whUpd1 + (size_t)l * 8192, upd_b1 + (size_t)l * 64,
            whUpd2 + (size_t)l * 4096, upd_b2 + (size_t)l * 64, N);
    }

    zero_kernel<<<1, 64, 0, stream>>>(out, NG);
    readout_kernel<<<(N + 63) / 64, 128, 0, stream>>>(
        xbuf, whEh1, eh_b1, eh_w2, eh_b2, batch, out, N);
}
